// LocalFeatureAggregation_8950711846141
// MI455X (gfx1250) — compile-verified, matrix operand order checked by reading, not run on hardware
//
#include <hip/hip_runtime.h>
#include <math.h>

typedef __attribute__((ext_vector_type(16))) _Float16 v16h;
typedef __attribute__((ext_vector_type(8)))  float    v8f;

union FragU {
    v16h h;
    unsigned int u[8];
};

// k-base for VGPR v of a 16x32 f16 A/B fragment (ISA 7.12.2):
// v in 0..3 -> k = 2*v + hi8 ; v in 4..7 -> k = 16 + 2*(v-4) + hi8
__device__ __forceinline__ int frag_kb(int v, int hi8) {
    return ((v & 3) << 1) + ((v & 4) << 2) + hi8;
}

// ---------------------------------------------------------------------------
// Generic tiled WMMA GEMM: C = act( A@W + bias [ + A2@W2 + bias2 ] )
// A: MxK fp32 row-major, W: KxNc fp32 row-major, C: MxNc fp32.
// K, K2 multiples of 32; Nc multiple of 64. act: 0=none 1=relu 2=leaky(0.01)
// Block: 256 threads (8 waves). BM=256, BN=64, BK=32.
// Wave tile: 32x64 = 2 A-frags x 4 B-frags = 8 WMMA per k-step.
// Double-buffered LDS; all fragments preloaded so DS waits are partial.
// ---------------------------------------------------------------------------
#define BM 256
#define BN 64
#define BK 32

__global__ __launch_bounds__(256)
void gemm_kernel(const float* __restrict__ A,  const float* __restrict__ W,
                 const float* __restrict__ bias,
                 const float* __restrict__ A2, const float* __restrict__ W2,
                 const float* __restrict__ bias2,
                 float* __restrict__ C,
                 int M, int K, int K2, int Nc, int act)
{
    __shared__ _Float16 As[2][BM * BK];   // [m][k] row-major, 2 x 16 KB
    __shared__ _Float16 Bs[2][BN * BK];   // W^T tile [n][k], 2 x 4 KB

    const int t    = threadIdx.x;
    const int lane = t & 31;
    const int w    = t >> 5;
    const int l15  = lane & 15;
    const int hi8  = (lane >> 4) << 3;
    const int m0   = blockIdx.x * BM;
    const int n0   = blockIdx.y * BN;

    v8f acc[2][4] = {};

    // --- staging: two 16-float A row-segments + two float4 W rows per thread ---
    auto load_regs = [&](const float* Ap, const float* Wp, int Kp, int k0,
                         float4* ar, float4* br) {
        int m = t >> 1, kof = (t & 1) << 4;
        #pragma unroll
        for (int h = 0; h < 2; ++h) {
            int row = m0 + m + h * 128; row = row < M ? row : M - 1;
            const float* ap = Ap + (size_t)row * Kp + k0 + kof;
            ar[h * 4 + 0] = *(const float4*)(ap + 0);
            ar[h * 4 + 1] = *(const float4*)(ap + 4);
            ar[h * 4 + 2] = *(const float4*)(ap + 8);
            ar[h * 4 + 3] = *(const float4*)(ap + 12);
        }
        int kk = t >> 4, nc4 = (t & 15) << 2;
        const float* wp = Wp + (size_t)(k0 + kk) * Nc + n0 + nc4;
        br[0] = *(const float4*)(wp);
        br[1] = *(const float4*)(wp + (size_t)16 * Nc);
    };

    // --- convert fp32->f16 and store into LDS buffer b ---
    auto store_tiles = [&](int b, const float4* ar, const float4* br) {
        int m = t >> 1, kof = (t & 1) << 4;
        #pragma unroll
        for (int h = 0; h < 2; ++h) {
            union { _Float16 hh[8]; uint4 q; } p0, p1;
            const float4* s = ar + h * 4;
            p0.hh[0] = (_Float16)s[0].x; p0.hh[1] = (_Float16)s[0].y;
            p0.hh[2] = (_Float16)s[0].z; p0.hh[3] = (_Float16)s[0].w;
            p0.hh[4] = (_Float16)s[1].x; p0.hh[5] = (_Float16)s[1].y;
            p0.hh[6] = (_Float16)s[1].z; p0.hh[7] = (_Float16)s[1].w;
            p1.hh[0] = (_Float16)s[2].x; p1.hh[1] = (_Float16)s[2].y;
            p1.hh[2] = (_Float16)s[2].z; p1.hh[3] = (_Float16)s[2].w;
            p1.hh[4] = (_Float16)s[3].x; p1.hh[5] = (_Float16)s[3].y;
            p1.hh[6] = (_Float16)s[3].z; p1.hh[7] = (_Float16)s[3].w;
            int mm = m + h * 128;
            *(uint4*)&As[b][mm * BK + kof]     = p0.q;
            *(uint4*)&As[b][mm * BK + kof + 8] = p1.q;
        }
        int kk = t >> 4, nc4 = (t & 15) << 2;
        Bs[b][(nc4 + 0) * BK + kk] = (_Float16)br[0].x;
        Bs[b][(nc4 + 1) * BK + kk] = (_Float16)br[0].y;
        Bs[b][(nc4 + 2) * BK + kk] = (_Float16)br[0].z;
        Bs[b][(nc4 + 3) * BK + kk] = (_Float16)br[0].w;
        Bs[b][(nc4 + 0) * BK + kk + 16] = (_Float16)br[1].x;
        Bs[b][(nc4 + 1) * BK + kk + 16] = (_Float16)br[1].y;
        Bs[b][(nc4 + 2) * BK + kk + 16] = (_Float16)br[1].z;
        Bs[b][(nc4 + 3) * BK + kk + 16] = (_Float16)br[1].w;
    };

    // preload ALL fragments into distinct registers, then chain 8 WMMAs so
    // the compiler can use partial s_wait_dscnt instead of full drains
    auto compute = [&](int b) {
        FragU afr[2];
        #pragma unroll
        for (int rt = 0; rt < 2; ++rt) {
            const _Float16* Ab = &As[b][(w * 32 + rt * 16 + l15) * BK];
            #pragma unroll
            for (int v = 0; v < 8; ++v)
                afr[rt].u[v] = *(const unsigned int*)(Ab + frag_kb(v, hi8));
        }
        FragU bfr[4];
        #pragma unroll
        for (int ct = 0; ct < 4; ++ct) {
            const _Float16* Bb = &Bs[b][(ct * 16 + l15) * BK];
            #pragma unroll
            for (int v = 0; v < 8; ++v)
                bfr[ct].u[v] = *(const unsigned int*)(Bb + frag_kb(v, hi8));
        }
        #pragma unroll
        for (int rt = 0; rt < 2; ++rt)
            #pragma unroll
            for (int ct = 0; ct < 4; ++ct)
                acc[rt][ct] = __builtin_amdgcn_wmma_f32_16x16x32_f16(
                    false, afr[rt].h, false, bfr[ct].h, (short)0, acc[rt][ct],
                    false, false);
    };

    int buf = 0;
    for (int ph = 0; ph < 2; ++ph) {
        const float* Ap = ph ? A2 : A;
        const float* Wp = ph ? W2 : W;
        const int    Kp = ph ? K2 : K;
        if (Ap == nullptr) break;          // uniform across the block

        float4 ar[8], br[2];
        load_regs(Ap, Wp, Kp, 0, ar, br);
        store_tiles(buf, ar, br);
        __syncthreads();

        for (int k0 = 0; k0 < Kp; k0 += BK) {
            bool more = (k0 + BK) < Kp;
            if (more) load_regs(Ap, Wp, Kp, k0 + BK, ar, br);  // overlap with WMMA
            compute(buf);
            if (more) store_tiles(buf ^ 1, ar, br);
            __syncthreads();
            if (more) buf ^= 1;
        }
    }

    // --- epilogue: bias(+bias2) + activation + store ---
    #pragma unroll
    for (int ct = 0; ct < 4; ++ct) {
        int n = n0 + ct * 16 + l15;
        float bv = bias[n] + (bias2 ? bias2[n] : 0.f);
        #pragma unroll
        for (int rt = 0; rt < 2; ++rt) {
            int mbase = m0 + w * 32 + rt * 16 + hi8;
            #pragma unroll
            for (int v = 0; v < 8; ++v) {
                int row = mbase + v;
                if (row < M) {
                    float x = acc[rt][ct][v] + bv;
                    if (act == 1)      x = fmaxf(x, 0.f);
                    else if (act == 2) x = x > 0.f ? x : 0.01f * x;
                    C[(size_t)row * Nc + n] = x;
                }
            }
        }
    }
}

// ---------------------------------------------------------------------------
// Fused LSE + attentive pool:
//   per point: gather K=16 neighbors, 10-d geometry built directly in the
//   WMMA A-fragment registers, MLP 10->64->128 (zero-padded K), concat
//   [enc(128) | x(128)], per-neighbor softmax over 256 channels, sum over K.
// Block: 256 threads = 8 waves = 16 points (one 16-row WMMA tile per point).
// Feature half handled analytically: Z_k = Ze_k + exp(mf-m_k)*Sf;
// feature output = x*exp(x-mf) * sum_k exp(mf-m_k)/Z_k.
// ---------------------------------------------------------------------------
__global__ __launch_bounds__(256)
void lse_pool_kernel(const float* __restrict__ coords, const int* __restrict__ nbr,
                     const float* __restrict__ xf,      // (N,128)
                     const float* __restrict__ W1, const float* __restrict__ b1, // 10x64
                     const float* __restrict__ W2, const float* __restrict__ b2, // 64x128
                     float* __restrict__ fout,          // (N,256)
                     int Npts)
{
    __shared__ _Float16 Wt1[64 * 32];        // [n][k] zero-padded K, 4 KB
    __shared__ _Float16 Wt2[128 * 64];       // [n][k], 16 KB
    __shared__ float    B1s[64];
    __shared__ float    B2s[128];
    __shared__ _Float16 Hst[8][16 * 64];     // per-wave hidden stage, 16 KB

    const int t = threadIdx.x;

    // ---- stage weights: batched loads into regs, then convert/store ----
    {
        float w1t[8];
        #pragma unroll
        for (int i = 0; i < 8; ++i) {
            int idx = t + i * 256; int n = idx >> 5, k = idx & 31;
            w1t[i] = (k < 10) ? W1[k * 64 + n] : 0.f;
        }
        #pragma unroll
        for (int i = 0; i < 8; ++i) Wt1[t + i * 256] = (_Float16)w1t[i];

        float4 w2t[8];
        int k8 = t >> 5, n4 = (t & 31) << 2;
        #pragma unroll
        for (int i = 0; i < 8; ++i)
            w2t[i] = *(const float4*)(W2 + (size_t)(k8 + i * 8) * 128 + n4);
        #pragma unroll
        for (int i = 0; i < 8; ++i) {
            int k = k8 + i * 8;
            Wt2[(n4 + 0) * 64 + k] = (_Float16)w2t[i].x;
            Wt2[(n4 + 1) * 64 + k] = (_Float16)w2t[i].y;
            Wt2[(n4 + 2) * 64 + k] = (_Float16)w2t[i].z;
            Wt2[(n4 + 3) * 64 + k] = (_Float16)w2t[i].w;
        }
        if (t < 64)  B1s[t] = b1[t];
        if (t < 128) B2s[t] = b2[t];
    }
    __syncthreads();

    const int w    = t >> 5;
    const int lane = t & 31;
    const int l15  = lane & 15;
    const int hi8  = (lane >> 4) << 3;

    for (int it = 0; it < 2; ++it) {
        int p = blockIdx.x * 16 + w * 2 + it;
        if (p >= Npts) p = Npts - 1;

        // ---- geometry for neighbor row m = l15 (both half-waves compute it) ----
        int nb = nbr[p * 16 + l15];
        float ox = coords[(size_t)p * 3 + 0], oy = coords[(size_t)p * 3 + 1], oz = coords[(size_t)p * 3 + 2];
        float nx = coords[(size_t)nb * 3 + 0], ny = coords[(size_t)nb * 3 + 1], nz = coords[(size_t)nb * 3 + 2];
        // feature half loads issued early (independent clause)
        float xr[4];
        #pragma unroll
        for (int i = 0; i < 4; ++i) xr[i] = xf[(size_t)p * 128 + lane + 32 * i];

        float rx = ox - nx, ry = oy - ny, rz = oz - nz;
        float d  = sqrtf(rx * rx + ry * ry + rz * rz);

        // A fragment (16x32, k>=10 zero): lanes<16 carry k=0..7, lanes>=16 k=8,9
        FragU a;
        #pragma unroll
        for (int i = 0; i < 8; ++i) a.u[i] = 0u;
        _Float16* ah = (_Float16*)&a;
        if (lane < 16) {
            ah[0] = (_Float16)ox; ah[1] = (_Float16)oy; ah[2] = (_Float16)oz;
            ah[3] = (_Float16)nx; ah[4] = (_Float16)ny; ah[5] = (_Float16)nz;
            ah[6] = (_Float16)rx; ah[7] = (_Float16)ry;
        } else {
            ah[0] = (_Float16)rz; ah[1] = (_Float16)d;
        }

        // ---- layer 1: (16x32) @ (32x64) ----
        v8f acc1[4] = {};
        #pragma unroll
        for (int ct = 0; ct < 4; ++ct) {
            FragU b;
            #pragma unroll
            for (int v = 0; v < 8; ++v)
                b.u[v] = *(const unsigned int*)(Wt1 + (ct * 16 + l15) * 32 + frag_kb(v, hi8));
            acc1[ct] = __builtin_amdgcn_wmma_f32_16x16x32_f16(
                false, a.h, false, b.h, (short)0, acc1[ct], false, false);
        }
        // relu + bias -> stage as f16
        #pragma unroll
        for (int ct = 0; ct < 4; ++ct) {
            int n = ct * 16 + l15;
            float bv = B1s[n];
            #pragma unroll
            for (int v = 0; v < 8; ++v) {
                float h = fmaxf(acc1[ct][v] + bv, 0.f);
                Hst[w][(v + hi8) * 64 + n] = (_Float16)h;
            }
        }
        __syncthreads();

        // ---- layer 2: (16x64) @ (64x128), 2 k-steps x 8 col tiles ----
        v8f acc2[8] = {};
        #pragma unroll
        for (int ks = 0; ks < 2; ++ks) {
            FragU a2;
            #pragma unroll
            for (int v = 0; v < 8; ++v)
                a2.u[v] = *(const unsigned int*)(&Hst[w][l15 * 64 + ks * 32 + frag_kb(v, hi8)]);
            #pragma unroll
            for (int ct = 0; ct < 8; ++ct) {
                FragU b;
                #pragma unroll
                for (int v = 0; v < 8; ++v)
                    b.u[v] = *(const unsigned int*)(Wt2 + (ct * 16 + l15) * 64 + ks * 32 + frag_kb(v, hi8));
                acc2[ct] = __builtin_amdgcn_wmma_f32_16x16x32_f16(
                    false, a2.h, false, b.h, (short)0, acc2[ct], false, false);
            }
        }
        #pragma unroll
        for (int ct = 0; ct < 8; ++ct) {
            float bv = B2s[ct * 16 + l15];
            #pragma unroll
            for (int v = 0; v < 8; ++v) acc2[ct][v] += bv;
        }

        // ---- attentive pooling ----
        float ef[4];
        float mf = fmaxf(fmaxf(xr[0], xr[1]), fmaxf(xr[2], xr[3]));
        #pragma unroll
        for (int s = 1; s < 32; s <<= 1) mf = fmaxf(mf, __shfl_xor(mf, s, 32));
        float Sf = 0.f;
        #pragma unroll
        for (int i = 0; i < 4; ++i) { ef[i] = __expf(xr[i] - mf); Sf += ef[i]; }
        #pragma unroll
        for (int s = 1; s < 32; s <<= 1) Sf += __shfl_xor(Sf, s, 32);

        // per-neighbor-row (k = v + hi8) stats over enc channels
        float mk[8], zinv[8], tkv[8];
        #pragma unroll
        for (int v = 0; v < 8; ++v) {
            float mx = acc2[0][v];
            #pragma unroll
            for (int ct = 1; ct < 8; ++ct) mx = fmaxf(mx, acc2[ct][v]);
            #pragma unroll
            for (int s = 1; s < 16; s <<= 1) mx = fmaxf(mx, __shfl_xor(mx, s, 32));
            mk[v] = fmaxf(mx, mf);
            float ze = 0.f;
            #pragma unroll
            for (int ct = 0; ct < 8; ++ct) ze += __expf(acc2[ct][v] - mk[v]);
            #pragma unroll
            for (int s = 1; s < 16; s <<= 1) ze += __shfl_xor(ze, s, 32);
            tkv[v] = __expf(mf - mk[v]);
            zinv[v] = 1.f / (ze + tkv[v] * Sf);
        }
        // enc-half contributions, summed over K
        float pe[8];
        #pragma unroll
        for (int ct = 0; ct < 8; ++ct) {
            float s = 0.f;
            #pragma unroll
            for (int v = 0; v < 8; ++v) {
                float val = acc2[ct][v];
                s += __expf(val - mk[v]) * zinv[v] * val;
            }
            pe[ct] = s + __shfl_xor(s, 16, 32);
        }
        float S = 0.f;
        #pragma unroll
        for (int v = 0; v < 8; ++v) S += tkv[v] * zinv[v];
        S += __shfl_xor(S, 16, 32);

        float* fp = fout + (size_t)p * 256;
        if (lane < 16) {
            #pragma unroll
            for (int ct = 0; ct < 8; ++ct) fp[ct * 16 + l15] = pe[ct];
        }
        #pragma unroll
        for (int i = 0; i < 4; ++i) fp[128 + lane + 32 * i] = xr[i] * ef[i] * S;
        __syncthreads();
    }
}

// ---------------------------------------------------------------------------
extern "C" void kernel_launch(void* const* d_in, const int* in_sizes, int n_in,
                              void* d_out, int out_size, void* d_ws, size_t ws_size,
                              hipStream_t stream)
{
    const float* coords = (const float*)d_in[0];
    const float* feats  = (const float*)d_in[1];
    const int*   nbr    = (const int*)d_in[2];
    const float *m1W1=(const float*)d_in[3],  *m1b1=(const float*)d_in[4],
                *m1W2=(const float*)d_in[5],  *m1b2=(const float*)d_in[6];
    const float *m2W1=(const float*)d_in[7],  *m2b1=(const float*)d_in[8],
                *m2W2=(const float*)d_in[9],  *m2b2=(const float*)d_in[10];
    const float *m3W1=(const float*)d_in[11], *m3b1=(const float*)d_in[12],
                *m3W2=(const float*)d_in[13], *m3b2=(const float*)d_in[14];
    const float *l1W1=(const float*)d_in[15], *l1b1=(const float*)d_in[16],
                *l1W2=(const float*)d_in[17], *l1b2=(const float*)d_in[18];
    const float *l2W1=(const float*)d_in[19], *l2b1=(const float*)d_in[20],
                *l2W2=(const float*)d_in[21], *l2b2=(const float*)d_in[22];
    const float *p1W1=(const float*)d_in[23], *p1b1=(const float*)d_in[24],
                *p1W2=(const float*)d_in[25], *p1b2=(const float*)d_in[26];
    const float *p2W1=(const float*)d_in[27], *p2b1=(const float*)d_in[28],
                *p2W2=(const float*)d_in[29], *p2b2=(const float*)d_in[30];
    float* out = (float*)d_out;

    const int N = in_sizes[2] / 16;   // 100000

    // workspace: s0 up to 384 cols, s1 up to 256 cols (N*640*4 ~= 256 MB)
    float* s0 = (float*)d_ws;
    float* s1 = s0 + (size_t)N * 384;

    dim3 blk(256);
    auto grid = [&](int M, int Nc) { return dim3((M + BM - 1) / BM, Nc / BN); };
    const float* np = nullptr;

    // m1: features -> h1 (relu) -> x1
    gemm_kernel<<<grid(N,128), blk, 0, stream>>>(feats, m1W1, m1b1, np,np,np, s0, N,128,0,128, 1);
    gemm_kernel<<<grid(N,128), blk, 0, stream>>>(s0,    m1W2, m1b2, np,np,np, s1, N,128,0,128, 0);
    // lse1 + pool -> f1
    lse_pool_kernel<<<(N + 15) / 16, blk, 0, stream>>>(coords, nbr, s1, l1W1,l1b1,l1W2,l1b2, s0, N);
    // p1: f1 -> (relu) -> x2
    gemm_kernel<<<grid(N,256), blk, 0, stream>>>(s0, p1W1, p1b1, np,np,np, s1, N,256,0,256, 1);
    gemm_kernel<<<grid(N,128), blk, 0, stream>>>(s1, p1W2, p1b2, np,np,np, s0, N,256,0,128, 0);
    // lse2 + pool -> f2
    lse_pool_kernel<<<(N + 15) / 16, blk, 0, stream>>>(coords, nbr, s0, l2W1,l2b1,l2W2,l2b2, s1, N);
    // p2: f2 -> (relu) -> x3
    gemm_kernel<<<grid(N,256), blk, 0, stream>>>(s1, p2W1, p2b1, np,np,np, s0, N,256,0,256, 1);
    gemm_kernel<<<grid(N,256), blk, 0, stream>>>(s0, p2W2, p2b2, np,np,np, s1, N,256,0,256, 0);
    // m2 hidden: x3 -> h2 (relu)   [s1 -> s0, width 384]
    gemm_kernel<<<grid(N,384), blk, 0, stream>>>(s1, m2W1, m2b1, np,np,np, s0, N,256,0,384, 1);
    // m3 hidden: features -> h3 (relu)   [-> s1, width 256]
    gemm_kernel<<<grid(N,256), blk, 0, stream>>>(feats, m3W1, m3b1, np,np,np, s1, N,128,0,256, 1);
    // fused final: out = leaky( h2@m2W2 + m2b2  +  h3@m3W2 + m3b2 )
    gemm_kernel<<<grid(N,512), blk, 0, stream>>>(s0, m2W2, m2b2, s1, m3W2, m3b2, out, N,384,256,512, 2);
}
